// LocationPrediction_RNN_67508295958743
// MI455X (gfx1250) — matrix operand order repaired and checked by reading, not compiled, not run on hardware
//
#include <hip/hip_runtime.h>
#include <hip/hip_bf16.h>

// MI455X / gfx1250 implementation of the flashback location-prediction forward
// pass. All GEMMs use v_wmma_f32_16x16x32_f16 (wave32 WMMA, f32 accumulate).
// The dominant FC GEMM ([8192x256]@[256x10000], 42 GFLOP) uses 5-wide N
// blocking per wave plus async global->LDS staging of the shared A tile
// (global_load_async_to_lds_b128 / s_wait_asynccnt, ASYNCcnt-tracked).

typedef __attribute__((ext_vector_type(16))) _Float16 v16h;
typedef __attribute__((ext_vector_type(8)))  float    v8f;

#define S_LEN 512
#define B_DIM 16
#define H_DIM 256
#define V_DIM 10000
#define SBROWS (S_LEN * B_DIM)          // 8192
#define NT_V   (V_DIM / 16)             // 625 = 125 * 5
#define DAY_F      86400.0f
#define LAM_T      0.1f
#define LAM_S      100.0f

#define WMMA_F16(acc, a, b) \
  (acc) = __builtin_amdgcn_wmma_f32_16x16x32_f16(false, (a), false, (b), (short)0, (acc), false, false)

// ---------------------------------------------------------------------------
// WMMA fragment loaders (f32 source -> f16 fragment), contiguous k order (k = 16h + s) in A and B.
// A: 16(M) x 32(K). Lane m = lane&15; K base 16 for lanes 16..31;
//    VGPR r holds K = kb + 2*r + {0,1}.
// ---------------------------------------------------------------------------
static __device__ __forceinline__ v16h load_a_f32(const float* a0, int lda, int lane) {
  const int m  = lane & 15;
  const int kb = (lane & 16) ? 16 : 0;
  const float* row = a0 + (size_t)m * lda;
  v16h a;
#pragma unroll
  for (int r = 0; r < 8; ++r) {
    const int k = kb + 2 * r;
    a[2 * r]     = (_Float16)row[k];
    a[2 * r + 1] = (_Float16)row[k + 1];
  }
  return a;
}

// B for C = A @ W^T, W row-major [N][K]: B[k][n] = W[n][k].
// Lane n = lane&15; K base 16 for lanes 16..31; VGPR r -> K = kb + 2r + {0,1}.
static __device__ __forceinline__ v16h load_bT_f32(const float* w0, int ldw, int lane) {
  const int n  = lane & 15;
  const int kb = (lane & 16) ? 16 : 0;
  const float* row = w0 + (size_t)n * ldw + kb;
  v16h b;
#pragma unroll
  for (int r = 0; r < 8; ++r) {
    b[2 * r]     = (_Float16)row[2 * r];
    b[2 * r + 1] = (_Float16)row[2 * r + 1];
  }
  return b;
}

// B row-major [K][N] (e.g. attention value matrix).
static __device__ __forceinline__ v16h load_bKN_f32(const float* b0, int ldb, int lane) {
  const int n  = lane & 15;
  const int kb = (lane & 16) ? 16 : 0;
  v16h b;
#pragma unroll
  for (int r = 0; r < 8; ++r) {
    const int k = kb + 2 * r;
    b[2 * r]     = (_Float16)b0[(size_t)k * ldb + n];
    b[2 * r + 1] = (_Float16)b0[(size_t)(k + 1) * ldb + n];
  }
  return b;
}

// ---------------------------------------------------------------------------
// K1: embedding gather  x_emb[s*B+b][h] = enc_emb[x[s,b]][h]
// ---------------------------------------------------------------------------
__global__ void k_embed(const int* __restrict__ x, const float* __restrict__ enc_emb,
                        float* __restrict__ xemb) {
  const int idx = blockIdx.x * 256 + threadIdx.x;   // SBROWS*H threads
  const int row = idx >> 8;                          // / H (H==256)
  const int hh  = idx & 255;
  const int tok = x[row];
  xemb[idx] = enc_emb[(size_t)tok * H_DIM + hh];
}

// ---------------------------------------------------------------------------
// K2: input projections Xf = x_emb @ Wih_f^T, Xb = x_emb @ Wih_b^T
// grid (2048, 2), block 128 (4 waves, 1 tile each). 8192 tiles per direction.
// ---------------------------------------------------------------------------
__global__ void k_xproj(const float* __restrict__ xemb,
                        const float* __restrict__ Wih_f, const float* __restrict__ Wih_b,
                        float* __restrict__ Xf, float* __restrict__ Xb) {
  const int dir  = blockIdx.y;
  const float* W = dir ? Wih_b : Wih_f;
  float* C       = dir ? Xb : Xf;
  const int wave = threadIdx.x >> 5, lane = threadIdx.x & 31;
  const int tile = blockIdx.x * 4 + wave;           // 512 (M) x 16 (N) tiles
  const int mt = tile >> 4, nt = tile & 15;
  const float* a_base = xemb + (size_t)mt * 16 * H_DIM;
  const float* w_base = W + (size_t)nt * 16 * H_DIM;
  v8f acc = {};
#pragma unroll
  for (int kk = 0; kk < H_DIM; kk += 32) {
    v16h a = load_a_f32(a_base + kk, H_DIM, lane);
    v16h b = load_bT_f32(w_base + kk, H_DIM, lane);
    WMMA_F16(acc, a, b);
  }
  const int n = lane & 15, mo = (lane & 16) ? 8 : 0;
  float* c0 = C + (size_t)(mt * 16) * H_DIM + nt * 16 + n;
#pragma unroll
  for (int r = 0; r < 8; ++r) c0[(size_t)(mo + r) * H_DIM] = acc[r];
}

// ---------------------------------------------------------------------------
// K3: sequential tanh RNN, both directions (grid.x = 2), 512 threads = 16 waves.
// Whh fragments register-cached across all 512 steps; h ping-pongs in LDS.
// ---------------------------------------------------------------------------
#define HSTR 264   // padded LDS row stride (floats) to spread banks
__global__ void k_rnn(const float* __restrict__ Xf, const float* __restrict__ Xb,
                      const float* __restrict__ Whh_f, const float* __restrict__ b_f,
                      const float* __restrict__ Whh_b, const float* __restrict__ b_b,
                      const float* __restrict__ h0,
                      float* __restrict__ out_f, float* __restrict__ hout) {
  __shared__ float hbuf[2][B_DIM * HSTR];
  const int dir     = blockIdx.x;
  const float* X    = dir ? Xb : Xf;
  const float* Whh  = dir ? Whh_b : Whh_f;
  const float* bias = dir ? b_b : b_f;
  const int tid = threadIdx.x;
  const int wave = tid >> 5, lane = tid & 31;

  for (int i = tid; i < B_DIM * H_DIM; i += 512) {
    const int b = i >> 8, hh = i & 255;
    hbuf[0][b * HSTR + hh] = h0[(size_t)dir * B_DIM * H_DIM + i];
  }
  // Register-cache Whh^T fragments for this wave's N tile (nt = wave).
  const int nt = wave;
  v16h wfrag[8];
#pragma unroll
  for (int kk = 0; kk < 8; ++kk)
    wfrag[kk] = load_bT_f32(Whh + (size_t)nt * 16 * H_DIM + kk * 32, H_DIM, lane);

  const int n = lane & 15, mo = (lane & 16) ? 8 : 0;
  const int nc = nt * 16 + n;
  const float bv = bias[nc];
  __syncthreads();

  int cur = 0;
  for (int t = 0; t < S_LEN; ++t) {
    const int si = dir ? (S_LEN - 1 - t) : t;
    const float* Xrow = X + (size_t)si * B_DIM * H_DIM;
    // Split K chain into two accumulators to halve WMMA dependency depth.
    v8f acc0 = {}, acc1 = {};
#pragma unroll
    for (int kk = 0; kk < 4; ++kk) {
      v16h a = load_a_f32(&hbuf[cur][kk * 32], HSTR, lane);
      WMMA_F16(acc0, a, wfrag[kk]);
    }
#pragma unroll
    for (int kk = 4; kk < 8; ++kk) {
      v16h a = load_a_f32(&hbuf[cur][kk * 32], HSTR, lane);
      WMMA_F16(acc1, a, wfrag[kk]);
    }
#pragma unroll
    for (int r = 0; r < 8; ++r) {
      const int m = mo + r;   // batch index
      const float v = tanhf(acc0[r] + acc1[r] + Xrow[m * H_DIM + nc] + bv);
      hbuf[cur ^ 1][m * HSTR + nc] = v;
      if (dir == 0) out_f[(size_t)t * B_DIM * H_DIM + m * H_DIM + nc] = v;
    }
    __syncthreads();
    cur ^= 1;
  }
  for (int i = tid; i < B_DIM * H_DIM; i += 512) {
    const int b = i >> 8, hh = i & 255;
    hout[(size_t)dir * B_DIM * H_DIM + i] = hbuf[cur][b * HSTR + hh];
  }
}

// ---------------------------------------------------------------------------
// K4: causal decay weights w[b][i][j] (f32) and row sums sum_w[i][b].
// ---------------------------------------------------------------------------
__global__ void k_weights(const float* __restrict__ t, const float* __restrict__ s,
                          float* __restrict__ w, float* __restrict__ sum_w) {
  const int i = blockIdx.x, b = blockIdx.y;
  __shared__ float red[256];
  const float ti  = t[i * B_DIM + b];
  const float si0 = s[(i * B_DIM + b) * 2 + 0];
  const float si1 = s[(i * B_DIM + b) * 2 + 1];
  float acc = 0.f;
  float* wrow = w + ((size_t)b * S_LEN + i) * S_LEN;
  for (int j = threadIdx.x; j < S_LEN; j += 256) {
    float val = 0.f;
    if (j <= i) {
      const float dt = ti - t[j * B_DIM + b];
      const float d0 = si0 - s[(j * B_DIM + b) * 2 + 0];
      const float d1 = si1 - s[(j * B_DIM + b) * 2 + 1];
      const float ds = sqrtf(d0 * d0 + d1 * d1);
      const float a  = (__cosf(dt * (6.28318530717958647692f / DAY_F)) + 1.f) * 0.5f
                     * __expf(-(dt / DAY_F) * LAM_T);
      val = a * __expf(-ds * LAM_S) + 1e-10f;
    }
    wrow[j] = val;
    acc += val;
  }
  red[threadIdx.x] = acc;
  __syncthreads();
  for (int st = 128; st > 0; st >>= 1) {
    if (threadIdx.x < st) red[threadIdx.x] += red[threadIdx.x + st];
    __syncthreads();
  }
  if (threadIdx.x == 0) sum_w[i * B_DIM + b] = red[0];
}

// ---------------------------------------------------------------------------
// K5: Q_b = (w_b @ out_f_b) / sum_w  -> Q[b][S][H] (f32)
// ---------------------------------------------------------------------------
__global__ void k_qgemm(const float* __restrict__ w, const float* __restrict__ outf,
                        const float* __restrict__ sum_w, float* __restrict__ Q) {
  const int b = blockIdx.y;
  const int wave = threadIdx.x >> 5, lane = threadIdx.x & 31;
  const int tile = blockIdx.x * 4 + wave;
  const int mt = tile >> 4, nt = tile & 15;
  const float* A  = w + (size_t)b * S_LEN * S_LEN + (size_t)mt * 16 * S_LEN;
  const float* Bm = outf + (size_t)b * H_DIM + nt * 16;  // [k][n] at k*(B*H)+n
  v8f acc = {};
  for (int kk = 0; kk < S_LEN; kk += 32) {
    v16h a  = load_a_f32(A + kk, S_LEN, lane);
    v16h bb = load_bKN_f32(Bm + (size_t)kk * B_DIM * H_DIM, B_DIM * H_DIM, lane);
    WMMA_F16(acc, a, bb);
  }
  const int n = lane & 15, mo = (lane & 16) ? 8 : 0;
  float* c0 = Q + ((size_t)b * S_LEN + mt * 16) * H_DIM + nt * 16 + n;
  const float* sw = sum_w + (mt * 16) * B_DIM + b;
#pragma unroll
  for (int r = 0; r < 8; ++r) {
    const int m = mo + r;
    c0[(size_t)m * H_DIM] = acc[r] / sw[m * B_DIM];
  }
}

// ---------------------------------------------------------------------------
// K6: scores_b = Q_b @ Q_b^T / sqrt(2H).
// ---------------------------------------------------------------------------
__global__ void k_scores(const float* __restrict__ Q, float* __restrict__ scores) {
  const int b = blockIdx.y;
  const int wave = threadIdx.x >> 5, lane = threadIdx.x & 31;
  const int tile = blockIdx.x * 4 + wave;           // 32 x 32 tiles
  const int mt = tile >> 5, nt = tile & 31;
  const float* Qb = Q + (size_t)b * S_LEN * H_DIM;
  const float* A = Qb + (size_t)mt * 16 * H_DIM;
  const float* W = Qb + (size_t)nt * 16 * H_DIM;
  v8f acc = {};
#pragma unroll
  for (int kk = 0; kk < H_DIM; kk += 32) {
    v16h a  = load_a_f32(A + kk, H_DIM, lane);
    v16h bb = load_bT_f32(W + kk, H_DIM, lane);
    WMMA_F16(acc, a, bb);
  }
  const float scale = 0.04419417382415922f;         // 1/sqrt(512)
  const int n = lane & 15, mo = (lane & 16) ? 8 : 0;
  float* c0 = scores + ((size_t)b * S_LEN + mt * 16) * S_LEN + nt * 16 + n;
#pragma unroll
  for (int r = 0; r < 8; ++r) c0[(size_t)(mo + r) * S_LEN] = acc[r] * scale;
}

// ---------------------------------------------------------------------------
// K7: row softmax in place over scores (8192 rows of 512).
// ---------------------------------------------------------------------------
__global__ void k_softmax(float* __restrict__ sc) {
  float* p = sc + (size_t)blockIdx.x * S_LEN;
  __shared__ float red[256];
  const int tid = threadIdx.x;
  const float v0 = p[tid], v1 = p[tid + 256];
  red[tid] = fmaxf(v0, v1);
  __syncthreads();
  for (int st = 128; st > 0; st >>= 1) {
    if (tid < st) red[tid] = fmaxf(red[tid], red[tid + st]);
    __syncthreads();
  }
  const float mx = red[0];
  __syncthreads();
  const float e0 = __expf(v0 - mx), e1 = __expf(v1 - mx);
  red[tid] = e0 + e1;
  __syncthreads();
  for (int st = 128; st > 0; st >>= 1) {
    if (tid < st) red[tid] += red[tid + st];
    __syncthreads();
  }
  const float inv = 1.f / red[0];
  p[tid] = e0 * inv;
  p[tid + 256] = e1 * inv;
}

// ---------------------------------------------------------------------------
// K8: ctx rows = alpha_b @ Q_b, scattered to ctx[s*B+b][H].
// ---------------------------------------------------------------------------
__global__ void k_ctx(const float* __restrict__ alpha, const float* __restrict__ Q,
                      float* __restrict__ ctx) {
  const int b = blockIdx.y;
  const int wave = threadIdx.x >> 5, lane = threadIdx.x & 31;
  const int tile = blockIdx.x * 4 + wave;           // 32 (M) x 16 (N)
  const int mt = tile >> 4, nt = tile & 15;
  const float* A  = alpha + (size_t)b * S_LEN * S_LEN + (size_t)mt * 16 * S_LEN;
  const float* Bm = Q + (size_t)b * S_LEN * H_DIM + nt * 16;  // [k][n] at k*H+n
  v8f acc = {};
  for (int kk = 0; kk < S_LEN; kk += 32) {
    v16h a  = load_a_f32(A + kk, S_LEN, lane);
    v16h bb = load_bKN_f32(Bm + (size_t)kk * H_DIM, H_DIM, lane);
    WMMA_F16(acc, a, bb);
  }
  const int n = lane & 15, mo = (lane & 16) ? 8 : 0;
  float* c0 = ctx + ((size_t)(mt * 16) * B_DIM + b) * H_DIM + nt * 16 + n;
#pragma unroll
  for (int r = 0; r < 8; ++r)
    c0[(size_t)(mo + r) * B_DIM * H_DIM] = acc[r];
}

// ---------------------------------------------------------------------------
// K9a: p_u gather.  K9b: userterm[b][v] = p_u @ fc_W[:,2H:]^T + fc_b.
// ---------------------------------------------------------------------------
__global__ void k_pu(const int* __restrict__ au, const float* __restrict__ user_emb,
                     float* __restrict__ pu) {
  const int b = blockIdx.x;
  const int uid = au[b];
  for (int hh = threadIdx.x; hh < H_DIM; hh += blockDim.x)
    pu[b * H_DIM + hh] = user_emb[(size_t)uid * H_DIM + hh];
}

__global__ void k_user(const float* __restrict__ pu, const float* __restrict__ fc_W,
                       const float* __restrict__ fc_b, float* __restrict__ uterm) {
  const int wave = threadIdx.x >> 5, lane = threadIdx.x & 31;
  const int nt = blockIdx.x * 4 + wave;
  if (nt >= NT_V) return;                            // wave-uniform guard
  const float* Wrow = fc_W + (size_t)nt * 16 * (3 * H_DIM) + 2 * H_DIM;
  v8f acc = {};
#pragma unroll
  for (int kk = 0; kk < H_DIM; kk += 32) {
    v16h a  = load_a_f32(pu + kk, H_DIM, lane);
    v16h bb = load_bT_f32(Wrow + kk, 3 * H_DIM, lane);
    WMMA_F16(acc, a, bb);
  }
  const int n = lane & 15, mo = (lane & 16) ? 8 : 0;
  const int vcol = nt * 16 + n;
  const float bv = fc_b[vcol];
#pragma unroll
  for (int r = 0; r < 8; ++r)
    uterm[(size_t)(mo + r) * V_DIM + vcol] = acc[r] + bv;
}

// ---------------------------------------------------------------------------
// K10: y[row][v] = ctx[row] @ fc_W[v, :H] + userterm[row%B][v]
// Block = 4 waves sharing one M tile; A tile (16x256 f32) async-staged into
// LDS via global_load_async_to_lds_b128 (ASYNCcnt), then each wave computes
// 5 N tiles (625 = 125*5), reusing each LDS A fragment 5x per K step.
// grid = 512 (mt) * 32 (n-group) = 16384 blocks, 128 threads.
// ---------------------------------------------------------------------------
#define AST 264    // padded LDS row stride (floats)
__global__ void k_fc(const float* __restrict__ ctx, const float* __restrict__ fc_W,
                     const float* __restrict__ uterm, float* __restrict__ y) {
  __shared__ float Atile[16 * AST];
  const int wave = threadIdx.x >> 5, lane = threadIdx.x & 31;
  const int mt = blockIdx.x >> 5;                   // 512 M tiles
  const int ng = blockIdx.x & 31;                   // 32 N groups
  const float* Abase = ctx + (size_t)mt * 16 * H_DIM;

  // Async-stage the 16x256 f32 A tile into LDS (8 x 128 lanes x 16B = 16 KB).
  {
    const int tid = threadIdx.x;
#pragma unroll
    for (int it = 0; it < 8; ++it) {
      const int idx  = it * 128 + tid;              // float4 index 0..1023
      const int row  = idx >> 6;                    // 64 float4 per source row
      const int col4 = idx & 63;
      const unsigned lds_off =
          (unsigned)(unsigned long long)(const void*)&Atile[row * AST + col4 * 4];
      const unsigned goff = (unsigned)(idx * 16);   // byte offset into A tile
      asm volatile("global_load_async_to_lds_b128 %0, %1, %2 offset:0"
                   :: "v"(lds_off), "v"(goff), "s"(Abase) : "memory");
    }
    asm volatile("s_wait_asynccnt 0" ::: "memory");
  }
  __syncthreads();

  const int quint = ng * 4 + wave;                  // 0..127; need < 125
  if (quint < 125) {                                // wave-uniform guard
    const int nt0 = quint * 5;
    const float* W0 = fc_W + (size_t)nt0 * 16 * (3 * H_DIM);
    __builtin_prefetch(W0, 0, 1);
    v8f acc[5] = {};
#pragma unroll
    for (int kk = 0; kk < H_DIM; kk += 32) {
      v16h a = load_a_f32(&Atile[kk], AST, lane);   // LDS reads (DScnt)
#pragma unroll
      for (int q = 0; q < 5; ++q) {
        v16h bb = load_bT_f32(W0 + (size_t)q * 16 * (3 * H_DIM) + kk, 3 * H_DIM, lane);
        WMMA_F16(acc[q], a, bb);
      }
    }
    const int n = lane & 15, mo = (lane & 16) ? 8 : 0;
#pragma unroll
    for (int q = 0; q < 5; ++q) {
      const int vcol = (nt0 + q) * 16 + n;
#pragma unroll
      for (int r = 0; r < 8; ++r) {
        const int row = mt * 16 + mo + r;           // row = s*B + b, b = row & 15
        y[(long)row * V_DIM + vcol] = acc[q][r] + uterm[(size_t)(row & 15) * V_DIM + vcol];
      }
    }
  }
}

// ---------------------------------------------------------------------------
// Launch: all on `stream`; workspace laid out with liveness-based aliasing.
//   region1 [0 .. 6291456): x_emb|Xf|Xb  ->  reused as w  ->  reused as scores
//   out_f 2097152 | sum_w 8192 | Q 2097152 | ctx 2097152 | uterm 160000 | pu 4096
//   total = 12,755,200 floats (~51 MB) of d_ws.
// ---------------------------------------------------------------------------
extern "C" void kernel_launch(void* const* d_in, const int* in_sizes, int n_in,
                              void* d_out, int out_size, void* d_ws, size_t ws_size,
                              hipStream_t stream) {
  (void)in_sizes; (void)n_in; (void)out_size; (void)ws_size;
  const int*   x        = (const int*)  d_in[0];
  const float* t        = (const float*)d_in[1];
  const float* s        = (const float*)d_in[2];
  const float* h0       = (const float*)d_in[5];
  const int*   au       = (const int*)  d_in[6];
  const float* enc_emb  = (const float*)d_in[7];
  const float* user_emb = (const float*)d_in[8];
  const float* Wih_f    = (const float*)d_in[9];
  const float* Whh_f    = (const float*)d_in[10];
  const float* b_f      = (const float*)d_in[11];
  const float* Wih_b    = (const float*)d_in[12];
  const float* Whh_b    = (const float*)d_in[13];
  const float* b_b      = (const float*)d_in[14];
  const float* fc_W     = (const float*)d_in[15];
  const float* fc_b     = (const float*)d_in[16];

  float* y    = (float*)d_out;
  float* hout = y + (size_t)SBROWS * V_DIM;

  float* ws    = (float*)d_ws;
  float* xemb  = ws + 0;
  float* Xf    = ws + 2097152;
  float* Xb    = ws + 4194304;
  float* wbuf  = ws + 0;        // alias: x_emb/Xf dead when written (after k_rnn)
  float* scmat = ws + 0;        // alias: wbuf dead when written (after k_qgemm)
  float* outf  = ws + 6291456;
  float* sumw  = ws + 8388608;
  float* Q     = ws + 8396800;
  float* ctx   = ws + 10493952;
  float* uterm = ws + 12591104;
  float* pu    = ws + 12751104;

  k_embed  <<<SBROWS * H_DIM / 256, 256, 0, stream>>>(x, enc_emb, xemb);
  k_xproj  <<<dim3(2048, 2), 128, 0, stream>>>(xemb, Wih_f, Wih_b, Xf, Xb);
  k_rnn    <<<2, 512, 0, stream>>>(Xf, Xb, Whh_f, b_f, Whh_b, b_b, h0, outf, hout);
  k_weights<<<dim3(S_LEN, B_DIM), 256, 0, stream>>>(t, s, wbuf, sumw);
  k_qgemm  <<<dim3(128, B_DIM), 128, 0, stream>>>(wbuf, outf, sumw, Q);
  k_scores <<<dim3(256, B_DIM), 128, 0, stream>>>(Q, scmat);
  k_softmax<<<SBROWS, 256, 0, stream>>>(scmat);
  k_ctx    <<<dim3(128, B_DIM), 128, 0, stream>>>(scmat, Q, ctx);
  k_pu     <<<B_DIM, 256, 0, stream>>>(au, user_emb, pu);
  k_user   <<<157, 128, 0, stream>>>(pu, fc_W, fc_b, uterm);
  k_fc     <<<dim3(512 * 32), 128, 0, stream>>>(ctx, fc_W, uterm, y);
}
